// multi_gcn_d_new_44727789421322
// MI455X (gfx1250) — matrix operand order repaired and checked by reading, not compiled, not run on hardware
//
#include <hip/hip_runtime.h>
#include <hip/hip_bf16.h>
#include <math.h>

// Problem constants (from reference): B=8, C=32, N=512, T=24, c_out=32
#define BT_ 192   // B*T
#define NN  512
#define CC  32
#define TT  24

typedef __attribute__((ext_vector_type(16))) __bf16 bf16x16;
typedef __attribute__((ext_vector_type(8)))  float  f32x8;
typedef __attribute__((ext_vector_type(4)))  unsigned int u32x4;
typedef __attribute__((ext_vector_type(8)))  unsigned int u32x8;

union FragB16 {
  bf16x16 v;
  unsigned int u32[8];
  unsigned short u16[16];
};

__device__ __forceinline__ unsigned short f2bf(float f) {
  union { float f; unsigned int u; } c; c.f = f;
  unsigned int r = c.u + 0x7FFFu + ((c.u >> 16) & 1u);   // round-to-nearest-even
  return (unsigned short)(r >> 16);
}
__device__ __forceinline__ float bf2f(unsigned short h) {
  union { float f; unsigned int u; } c; c.u = ((unsigned int)h) << 16;
  return c.f;
}

__device__ __forceinline__ f32x8 wmma_bf16(bf16x16 a, bf16x16 b, f32x8 c) {
  // D = A(16x32 bf16) * B(32x16 bf16) + C(16x16 f32)
  return __builtin_amdgcn_wmma_f32_16x16x32_bf16(false, a, false, b, (short)0, c,
                                                 false, false);
}

// 16-bit A-matrix 16x32 VGPR layout: lane group g (=lane/16), vgpr v holds K,K+1
__device__ __forceinline__ int kA(int v, int g) {
  return ((v >> 2) << 4) + (g << 3) + ((v & 3) << 1);
}
// 16-bit B-matrix 32x16 VGPR layout
__device__ __forceinline__ int kB(int v, int g) { return ((v >> 2) << 4) + (g << 3) + ((v & 3) << 1); }

// ---------------------------------------------------------------------------
// Tensor Data Mover: 1-D copy of n2 bf16 elements from global to LDS offset 0.
// D# per CDNA5 ISA ch.8: group0 = {count=1, lds_addr, global_addr, type=2},
// group1 = {data_size=2B, tensor_dim0=tile_dim0=n2, tensor_dim1=tile_dim1=1,
//           tensor_dim0_stride=n2}.  Issue from ONE wave; tracked by TENSORcnt.
// ---------------------------------------------------------------------------
__device__ __forceinline__ void tdm_load_lds0(const void* gsrc, unsigned n2) {
  unsigned long long ga = (unsigned long long)gsrc;
  u32x4 g0;
  g0[0] = 1u;                                    // count=1 valid user descriptor
  g0[1] = 0u;                                    // lds_addr = 0
  g0[2] = (unsigned)ga;                          // global_addr[31:0]
  g0[3] = (unsigned)((ga >> 32) & 0x01FFFFFFu) | (2u << 30); // addr[56:32]|type=2
  u32x8 g1;
  g1[0] = 1u << 16;                              // wg_mask=0, data_size=1 (2B)
  g1[1] = (n2 & 0xFFFFu) << 16;                  // tensor_dim0[15:0]
  g1[2] = (n2 >> 16) | (1u << 16);               // tensor_dim0[31:16] | tensor_dim1=1
  g1[3] = (n2 & 0xFFFFu) << 16;                  // tensor_dim1 hi=0 | tile_dim0=n2
  g1[4] = 1u;                                    // tile_dim1=1, tile_dim2=0
  g1[5] = n2;                                    // tensor_dim0_stride[31:0]
  g1[6] = 0u;
  g1[7] = 0u;
  asm volatile("tensor_load_to_lds %0, %1" :: "s"(g0), "s"(g1) : "memory");
}
__device__ __forceinline__ void tdm_wait() {
  __builtin_amdgcn_s_wait_tensorcnt(0);
}

// ---------------------------------------------------------------------------
// Kernel 0a: x [B,C,N,T] f32  ->  xt bf16 [bt][n][c]
// ---------------------------------------------------------------------------
__global__ void k_xt(const float* __restrict__ x, unsigned short* __restrict__ xt) {
  int idx = blockIdx.x * 256 + threadIdx.x;
  const int total = 8 * TT * NN * CC;
  if (idx >= total) return;
  int c = idx & (CC - 1);
  int n = (idx >> 5) & (NN - 1);
  int t = (idx >> 14) % TT;
  int b = idx >> 14; b /= TT;
  float v = x[(((size_t)b * CC + c) * NN + n) * TT + t];
  xt[idx] = f2bf(v);
}

// Kernel 0b: generic f32 -> bf16
__global__ void k_f2bf(const float* __restrict__ src, unsigned short* __restrict__ dst, int n) {
  int idx = blockIdx.x * 256 + threadIdx.x;
  if (idx < n) dst[idx] = f2bf(src[idx]);
}

// ---------------------------------------------------------------------------
// Kernel 1: per (b,t) attention + first-hop GCN.
// grid = BT*4 workgroups, 256 threads (8 waves); wave owns 16 attention rows.
// LDS: X tile (32KB bf16, offset 0, TDM-filled) + per-wave 16x512 f32 exp
// buffer (256KB) + reductions.  Writes x1_s (bf16) at h[2s].
// ---------------------------------------------------------------------------
__global__ void k_attn(const unsigned short* __restrict__ xt,
                       const unsigned short* __restrict__ abf,
                       unsigned short* __restrict__ h) {
  extern __shared__ char smem[];
  unsigned short* Xs = (unsigned short*)smem;                         // 512*32 bf16 @0
  float* expBuf = (float*)(smem + NN * CC * 2);                       // 8*16*512 f32
  float* pmax   = (float*)(smem + NN * CC * 2 + 8 * 16 * NN * 4);     // [8][32]
  float* psum   = pmax + 256;                                         // [8][32]
  float* rowInv = psum + 256;                                         // [8][16]

  const int bt   = blockIdx.x >> 2;
  const int blk  = blockIdx.x & 3;
  const int wave = threadIdx.x >> 5;
  const int lane = threadIdx.x & 31;
  const int m    = lane & 15;
  const int g    = lane >> 4;
  const int rowBase = blk * 128 + wave * 16;

  // TDM fill of the X tile: one wave issues, waits TENSORcnt, then all sync.
  if (wave == 0) {
    tdm_load_lds0(xt + (size_t)bt * NN * CC, NN * CC);
    tdm_wait();
  }
  __syncthreads();

  // A fragment: rows rowBase..+15 of X, K = C = 32 (single chunk)
  FragB16 aX;
#pragma unroll
  for (int v = 0; v < 8; ++v)
    aX.u32[v] = *(const unsigned int*)(Xs + (rowBase + m) * CC + kA(v, g));

  float* eb = expBuf + wave * (16 * NN);

  // score tiles: S[16 x 512] = X_rows * X^T, scaled by 1/sqrt(C)
  for (int nt = 0; nt < NN / 16; ++nt) {
    FragB16 bX;   // B[k][n] = X[nt*16+n][k]  -> contiguous bf16 pairs
#pragma unroll
    for (int v = 0; v < 8; ++v)
      bX.u32[v] = *(const unsigned int*)(Xs + (nt * 16 + m) * CC + kB(v, g));
    f32x8 cf = {};
    cf = wmma_bf16(aX.v, bX.v, cf);
#pragma unroll
    for (int r = 0; r < 8; ++r)
      eb[(r + 8 * g) * NN + nt * 16 + m] = cf[r] * 0.17677669529663689f; // 1/sqrt(32)
  }
  __syncthreads();

  // softmax per row; 2 lanes per row (halves of 256 columns)
  {
    float* rowp = eb + m * NN + g * 256;
    float mx = -3.4e38f;
    for (int j = 0; j < 256; ++j) mx = fmaxf(mx, rowp[j]);
    pmax[wave * 32 + lane] = mx;
    __syncthreads();
    mx = fmaxf(pmax[wave * 32 + m], pmax[wave * 32 + 16 + m]);
    float sum = 0.f;
    for (int j = 0; j < 256; ++j) {
      float e = __expf(rowp[j] - mx);
      rowp[j] = e;
      sum += e;
    }
    psum[wave * 32 + lane] = sum;
    __syncthreads();
    sum = psum[wave * 32 + m] + psum[wave * 32 + 16 + m];
    if (g == 0) rowInv[wave * 16 + m] = 1.0f / sum;
  }
  __syncthreads();

  // x1_s = (a_s o att) * X for all 3 supports, K-chunk outermost so the
  // X B-fragments and softmax probabilities are loaded once per chunk.
  const unsigned short* arow[3];
#pragma unroll
  for (int s = 0; s < 3; ++s)
    arow[s] = abf + (size_t)s * NN * NN + (size_t)(rowBase + m) * NN;
  const float* erow = eb + m * NN;
  const float rinv = rowInv[wave * 16 + m];

  f32x8 acc[3][2] = {};
  for (int kc = 0; kc < NN / 32; ++kc) {
    float e[16];
#pragma unroll
    for (int v = 0; v < 8; ++v) {
      int kk = kc * 32 + kA(v, g);
      e[2 * v]     = erow[kk]     * rinv;
      e[2 * v + 1] = erow[kk + 1] * rinv;
    }
    FragB16 b0, b1; // B[k][c] = X[kc*32+k][c], two 16-col tiles (c0=0,16)
#pragma unroll
    for (int v = 0; v < 8; ++v) {
      int kk = kc * 32 + kB(v, g);
      b0.u16[2 * v]     = Xs[kk * CC + m];
      b0.u16[2 * v + 1] = Xs[(kk + 1) * CC + m];
      b1.u16[2 * v]     = Xs[kk * CC + 16 + m];
      b1.u16[2 * v + 1] = Xs[(kk + 1) * CC + 16 + m];
    }
#pragma unroll
    for (int s = 0; s < 3; ++s) {
      FragB16 pa;     // P = a * softmax, built on the fly as bf16
#pragma unroll
      for (int v = 0; v < 8; ++v) {
        int kk = kc * 32 + kA(v, g);
        unsigned int au = *(const unsigned int*)(arow[s] + kk);
        pa.u16[2 * v]     = f2bf(e[2 * v]     * bf2f((unsigned short)au));
        pa.u16[2 * v + 1] = f2bf(e[2 * v + 1] * bf2f((unsigned short)(au >> 16)));
      }
      acc[s][0] = wmma_bf16(pa.v, b0.v, acc[s][0]);
      acc[s][1] = wmma_bf16(pa.v, b1.v, acc[s][1]);
    }
  }
#pragma unroll
  for (int s = 0; s < 3; ++s) {
    unsigned short* dst = h + ((size_t)(2 * s) * BT_ + bt) * NN * CC;
#pragma unroll
    for (int r = 0; r < 8; ++r) {
      int mr = rowBase + r + 8 * g;
      dst[mr * CC + m]      = f2bf(acc[s][0][r]);
      dst[mr * CC + 16 + m] = f2bf(acc[s][1][r]);
    }
  }
}

// ---------------------------------------------------------------------------
// Kernel 2: x2_s = a_s * x1_s  per (b,t,s).  grid = BT*3*4, 256 threads.
// ---------------------------------------------------------------------------
__global__ void k_hop2(const unsigned short* __restrict__ abf,
                       unsigned short* __restrict__ h) {
  __shared__ unsigned short Bs[NN * CC];   // LDS offset 0
  const int bt   = blockIdx.x / 12;
  const int rem  = blockIdx.x % 12;
  const int s    = rem >> 2;
  const int blk  = rem & 3;
  const int wave = threadIdx.x >> 5;
  const int lane = threadIdx.x & 31;
  const int m    = lane & 15;
  const int g    = lane >> 4;
  const int rowBase = blk * 128 + wave * 16;

  if (wave == 0) {
    tdm_load_lds0(h + ((size_t)(2 * s) * BT_ + bt) * NN * CC, NN * CC);
    tdm_wait();
  }
  __syncthreads();

  const unsigned short* Arow =
      abf + (size_t)s * NN * NN + (size_t)(rowBase + m) * NN;
  f32x8 acc0 = {}, acc1 = {};
  for (int kc = 0; kc < NN / 32; ++kc) {
    FragB16 pa;   // bf16 support rows, contiguous 32-bit loads
#pragma unroll
    for (int v = 0; v < 8; ++v)
      pa.u32[v] = *(const unsigned int*)(Arow + kc * 32 + kA(v, g));
    FragB16 b0, b1;
#pragma unroll
    for (int v = 0; v < 8; ++v) {
      int kk = kc * 32 + kB(v, g);
      b0.u16[2 * v]     = Bs[kk * CC + m];
      b0.u16[2 * v + 1] = Bs[(kk + 1) * CC + m];
      b1.u16[2 * v]     = Bs[kk * CC + 16 + m];
      b1.u16[2 * v + 1] = Bs[(kk + 1) * CC + 16 + m];
    }
    acc0 = wmma_bf16(pa.v, b0.v, acc0);
    acc1 = wmma_bf16(pa.v, b1.v, acc1);
  }
  unsigned short* dst = h + ((size_t)(2 * s + 1) * BT_ + bt) * NN * CC;
#pragma unroll
  for (int r = 0; r < 8; ++r) {
    int mr = rowBase + r + 8 * g;
    dst[mr * CC + m]      = f2bf(acc0[r]);
    dst[mr * CC + 16 + m] = f2bf(acc1[r]);
  }
}

// ---------------------------------------------------------------------------
// Kernel 3: y[b,:,n,t] = W[32x224] * h[224x512] + bias.  grid = BT, 256 threads.
// h channel chunks: [x, x1_0, x2_0, x1_1, x2_1, x1_2, x2_2], all [n][c] bf16.
// ---------------------------------------------------------------------------
__global__ void k_conv(const unsigned short* __restrict__ xtb,
                       const unsigned short* __restrict__ h,
                       const unsigned short* __restrict__ Wb,
                       const float* __restrict__ bias,
                       float* __restrict__ y) {
  __shared__ unsigned short Ws_[32 * 224];  // LDS offset 0
  const int bt   = blockIdx.x;
  const int b    = bt / TT, t = bt % TT;
  const int wave = threadIdx.x >> 5;
  const int lane = threadIdx.x & 31;
  const int m    = lane & 15;
  const int g    = lane >> 4;

  if (wave == 0) {
    tdm_load_lds0(Wb, 32 * 224);
    tdm_wait();
  }
  __syncthreads();

  FragB16 af[2][7];   // W row tiles (o=0..15, 16..31) x 7 K-chunks
#pragma unroll
  for (int mt = 0; mt < 2; ++mt)
#pragma unroll
    for (int ch = 0; ch < 7; ++ch)
#pragma unroll
      for (int v = 0; v < 8; ++v)
        af[mt][ch].u32[v] =
            *(const unsigned int*)(Ws_ + (mt * 16 + m) * 224 + ch * 32 + kA(v, g));

  const size_t tileOff = (size_t)bt * NN * CC;
  const unsigned short* srcs[7] = {
      xtb + tileOff,
      h + (size_t)0 * BT_ * NN * CC + tileOff,
      h + (size_t)1 * BT_ * NN * CC + tileOff,
      h + (size_t)2 * BT_ * NN * CC + tileOff,
      h + (size_t)3 * BT_ * NN * CC + tileOff,
      h + (size_t)4 * BT_ * NN * CC + tileOff,
      h + (size_t)5 * BT_ * NN * CC + tileOff};

  for (int i = 0; i < 4; ++i) {          // each wave: 4 column tiles of 16 points
    int n0 = (wave * 4 + i) * 16;
    f32x8 acc0 = {}, acc1 = {};
#pragma unroll
    for (int ch = 0; ch < 7; ++ch) {
      FragB16 bf;   // B[k][n] = src[n][k]: contiguous channel pairs
#pragma unroll
      for (int v = 0; v < 8; ++v)
        bf.u32[v] = *(const unsigned int*)(srcs[ch] + (n0 + m) * CC + kB(v, g));
      acc0 = wmma_bf16(af[0][ch].v, bf.v, acc0);
      acc1 = wmma_bf16(af[1][ch].v, bf.v, acc1);
    }
#pragma unroll
    for (int r = 0; r < 8; ++r) {
      int o0 = r + 8 * g;
      int n  = n0 + m;
      y[(((size_t)b * 32 + o0) * NN + n) * TT + t]      = acc0[r] + bias[o0];
      y[(((size_t)b * 32 + o0 + 16) * NN + n) * TT + t] = acc1[r] + bias[o0 + 16];
    }
  }
}

// ---------------------------------------------------------------------------
extern "C" void kernel_launch(void* const* d_in, const int* in_sizes, int n_in,
                              void* d_out, int out_size, void* d_ws, size_t ws_size,
                              hipStream_t stream) {
  const float* x    = (const float*)d_in[0];
  const float* s0   = (const float*)d_in[1];
  const float* s1   = (const float*)d_in[2];
  const float* s2   = (const float*)d_in[3];
  const float* W    = (const float*)d_in[4];
  const float* bias = (const float*)d_in[5];
  float* y = (float*)d_out;

  // workspace layout (bf16 = unsigned short), total ~45.6 MB:
  //   xtb [BT][N][C] | h: 6 x [BT][N][C] (x1_0,x2_0,x1_1,x2_1,x1_2,x2_2)
  //   abf: 3 x [N][N] | Wb: [32][224]
  const size_t tot = (size_t)BT_ * NN * CC;
  unsigned short* xtb = (unsigned short*)d_ws;
  unsigned short* h   = xtb + tot;
  unsigned short* abf = h + 6 * tot;
  unsigned short* Wb  = abf + 3 * (size_t)NN * NN;

  k_xt<<<(int)((tot + 255) / 256), 256, 0, stream>>>(x, xtb);
  k_f2bf<<<(NN * NN + 255) / 256, 256, 0, stream>>>(s0, abf + 0 * NN * NN, NN * NN);
  k_f2bf<<<(NN * NN + 255) / 256, 256, 0, stream>>>(s1, abf + 1 * NN * NN, NN * NN);
  k_f2bf<<<(NN * NN + 255) / 256, 256, 0, stream>>>(s2, abf + 2 * NN * NN, NN * NN);
  k_f2bf<<<(32 * 224 + 255) / 256, 256, 0, stream>>>(W, Wb, 32 * 224);

  // 32KB X tile + 256KB exp buffer + 2.5KB reductions = 297,472 B (<320KB WGP LDS)
  const size_t smem1 = (size_t)NN * CC * 2 + (size_t)8 * 16 * NN * 4 + (256 + 256 + 128) * 4;
  k_attn<<<BT_ * 4, 256, smem1, stream>>>(xtb, abf, h);
  k_hop2<<<BT_ * 12, 256, 0, stream>>>(abf, h);
  k_conv<<<BT_, 256, 0, stream>>>(xtb, h, Wb, bias, y);
}
